// DifferentiableGrammar_74165495267447
// MI455X (gfx1250) — matrix operand order repaired and checked by reading, not compiled, not run on hardware
//
#include <hip/hip_runtime.h>
#include <hip/hip_fp16.h>

typedef __attribute__((ext_vector_type(16))) _Float16 v16h;
typedef __attribute__((ext_vector_type(8)))  float    v8f;

#define N_SYM   12
#define T_STEPS 12
#define HDIM    100
#define BATCH   16384
#define G4      (4 * HDIM)       // 400 gate pre-activations
#define NTILES  25               // 25 M-tiles of 16 permuted gate rows
#define THREADS (NTILES * 32)    // 25 wave32 -> 800 threads, one WGP

#define LOG2E  1.4426950408889634f

// branch-free, saturating activations on native v_exp_f32 (base-2) + v_rcp_f32
__device__ __forceinline__ float fast_sigmoid(float x) {
    const float e = __builtin_amdgcn_exp2f(-LOG2E * x);
    return __builtin_amdgcn_rcpf(1.f + e);      // x->-inf: rcp(inf)=0 ; x->+inf: 1
}
__device__ __forceinline__ float fast_tanh(float x) {
    const float e = __builtin_amdgcn_exp2f(2.f * LOG2E * x);
    return 1.f - 2.f * __builtin_amdgcn_rcpf(1.f + e);  // e=inf -> 1 ; e=0 -> -1
}

// ---------------------------------------------------------------------------
// Phase 1: grammar expansion. one_hot @ grammar_mat is a row-gather (one-hot),
// so each batch element is an independent 11-step gumbel-argmax random walk.
// Only sequence[T-1] feeds LSTM row m=11 (the only row the reference reads),
// so we store just the final symbol per element, as a byte.
// ---------------------------------------------------------------------------
__device__ __forceinline__ unsigned pcg_hash(unsigned x) {
    x = x * 747796405u + 2891336453u;
    x = ((x >> ((x >> 28u) + 4u)) ^ x) * 277803737u;
    return (x >> 22u) ^ x;
}

__global__ void grammar_expand_kernel(const float* __restrict__ one_hot,
                                      const float* __restrict__ gmat,
                                      unsigned char* __restrict__ final_sym) {
    __shared__ float g[N_SYM * N_SYM];
    const int tid = threadIdx.x;
    if (tid < N_SYM * N_SYM) g[tid] = gmat[tid];
    __syncthreads();

    const int b = blockIdx.x * blockDim.x + tid;
    if (b >= BATCH) return;

    int s = 0;
    #pragma unroll
    for (int n = 0; n < N_SYM; ++n)
        if (one_hot[b * N_SYM + n] > 0.5f) s = n;

    for (int t = 1; t < T_STEPS; ++t) {
        if (s != N_SYM - 1) {
            float best = -1e30f;
            int   arg  = 0;
            #pragma unroll
            for (int n = 0; n < N_SYM; ++n) {
                unsigned r = pcg_hash(((unsigned)b * 131u + (unsigned)t) * 16u + (unsigned)n);
                float u    = (float)(r | 1u) * 2.3283064e-10f;     // (0,1]
                float gum  = -__logf(-__logf(u));                  // gumbel noise
                float v    = g[s * N_SYM + n] + gum;
                if (v > best) { best = v; arg = n; }
            }
            s = arg;
        }
    }
    final_sym[b] = (unsigned char)s;
}

// ---------------------------------------------------------------------------
// Phase 2: single persistent workgroup runs the serial LSTM chain (row m=11):
// 16384 dependent steps of gates[400] = W_hh * h + W_ih[:, sym].
//   - W_hh rows are PERMUTED into A-fragments so tile w holds rows
//     { t*100 + w*4 + j : t in 0..3 (i,f,g,o), j in 0..3 } -> all four gates
//     of h-indices w*4..w*4+3 stay inside one wave. One barrier per step.
//   - h replicated across all 16 B columns -> B addresses depend only on
//     lane>>4: flat 128-half h buffer, broadcast ds_load_b128, double-buffered.
//   - i,f (lanes 0-15) meet g,o (lanes 16-31) via two __shfl_xor(16).
//   - all 16384 input symbols + transposed W_ih resident in LDS.
//   - branch-free native-exp activations (no libm tanh branches).
// ---------------------------------------------------------------------------
__global__ void __launch_bounds__(THREADS, 1)
lstm_scan_kernel(const unsigned char* __restrict__ seq8,  // [BATCH]
                 const float* __restrict__ W_ih,          // [400, 12] row-major
                 const float* __restrict__ W_hh,          // [400, 100] row-major
                 float*       __restrict__ out) {         // [BATCH, 100]
    __shared__ _Float16 h_lds[2 * 128];         // double-buffered h, K padded
    __shared__ float    wih_t[N_SYM * G4];      // W_ih transposed: [sym][gate row]
    __shared__ unsigned sym_words[BATCH / 4];   // all input symbols, packed bytes

    const int tid  = threadIdx.x;
    const int lane = tid & 31;
    const int wave = tid >> 5;                  // 0..24 -> h-indices wave*4..wave*4+3

    // one-time staging (off the serial path)
    for (int i = tid; i < N_SYM * G4; i += THREADS) {
        const int n = i / G4, gidx = i % G4;
        wih_t[i] = W_ih[gidx * N_SYM + n];
    }
    for (int i = tid; i < BATCH / 4; i += THREADS)
        sym_words[i] = ((const unsigned*)seq8)[i];
    for (int i = tid; i < 2 * 128; i += THREADS)
        h_lds[i] = (_Float16)0.f;               // h0 = 0, pad K 100..127 = 0

    // --- preload A = permuted W_hh rows as 4 f16 fragments per wave ---
    // M position q (0..15) of tile w  ->  gate row (q>>2)*100 + w*4 + (q&3)
    // 16-bit A 16x32 K layout (contiguous, matches B loads):
    //   lanes 0-15  : halfs 0-15 -> K+0..15
    //   lanes 16-31 : halfs 0-15 -> K+16..31
    const int q    = lane & 15;
    const int row  = (q >> 2) * HDIM + wave * 4 + (q & 3);
    const int kofs = (lane >> 4) ? 16 : 0;
    v16h a_frag[4];
    #pragma unroll
    for (int kk = 0; kk < 4; ++kk) {
        #pragma unroll
        for (int j = 0; j < 16; ++j) {
            int   k = kk * 32 + j + kofs;
            float w = (k < HDIM) ? W_hh[row * HDIM + k] : 0.f;
            a_frag[kk][j] = (_Float16)w;
        }
    }

    // per-lane W_ih rows: lanes 0-15 provide i,f; lanes 16-31 provide g,o
    const int jl    = lane & 3;
    const int thalf = (lane >> 4) ? 2 : 0;
    const int r_lo  = (thalf + 0) * HDIM + wave * 4 + jl;   // i or g row
    const int r_hi  = (thalf + 1) * HDIM + wave * 4 + jl;   // f or o row

    const int  hb   = (lane >> 4) * 16;         // B K-slab base within 32-K block
    const bool actv = (lane < 4);               // activation lanes (4 h per wave)
    float c_state = 0.f;
    __syncthreads();

    const unsigned char* sym_bytes = (const unsigned char*)sym_words;

    for (int b = 0; b < BATCH; ++b) {
        const int cur = (b & 1) * 128;          // read buffer
        const int nxt = 128 - cur;              // write buffer

        const int   sym  = (int)sym_bytes[b];
        const float w_lo = wih_t[sym * G4 + r_lo];
        const float w_hi = wih_t[sym * G4 + r_hi];

        // gates_tile = W_hh_tile[16x128] * h (replicated over 16 columns):
        // two overlapped 2-deep WMMA chains
        const v16h bf0 = *(const v16h*)&h_lds[cur + hb];
        const v16h bf2 = *(const v16h*)&h_lds[cur + hb + 64];
        v8f d0 = {}, d1 = {};
        d0 = __builtin_amdgcn_wmma_f32_16x16x32_f16(false, a_frag[0], false, bf0,
                                                    (short)0, d0, false, false);
        d1 = __builtin_amdgcn_wmma_f32_16x16x32_f16(false, a_frag[2], false, bf2,
                                                    (short)0, d1, false, false);
        const v16h bf1 = *(const v16h*)&h_lds[cur + hb + 32];
        const v16h bf3 = *(const v16h*)&h_lds[cur + hb + 96];
        d0 = __builtin_amdgcn_wmma_f32_16x16x32_f16(false, a_frag[1], false, bf1,
                                                    (short)0, d0, false, false);
        d1 = __builtin_amdgcn_wmma_f32_16x16x32_f16(false, a_frag[3], false, bf3,
                                                    (short)0, d1, false, false);

        // lanes 0-15 hold tile rows 0-7 (i,f), lanes 16-31 rows 8-15 (g,o); columns equal.
        // diagonal select: lane jl takes element jl (i/g) and 4+jl (f/o).
        const float eA0 = d0[0] + d1[0], eA1 = d0[1] + d1[1];
        const float eA2 = d0[2] + d1[2], eA3 = d0[3] + d1[3];
        const float eB0 = d0[4] + d1[4], eB1 = d0[5] + d1[5];
        const float eB2 = d0[6] + d1[6], eB3 = d0[7] + d1[7];
        const float tA0 = (jl & 1) ? eA1 : eA0, tA1 = (jl & 1) ? eA3 : eA2;
        const float tB0 = (jl & 1) ? eB1 : eB0, tB1 = (jl & 1) ? eB3 : eB2;
        const float vA  = ((jl & 2) ? tA1 : tA0) + w_lo;    // i (lanes<16) / g (>=16)
        const float vB  = ((jl & 2) ? tB1 : tB0) + w_hi;    // f (lanes<16) / o (>=16)

        // bring g,o (upper half-wave) to lanes 0-3
        const float gA = __shfl_xor(vA, 16, 32);
        const float gB = __shfl_xor(vB, 16, 32);

        if (actv) {                             // lanes 0-3: 4 h-values per wave
            const float si = fast_sigmoid(vA);
            const float sf = fast_sigmoid(vB);
            const float so = fast_sigmoid(gB);
            c_state = sf * c_state + si * fast_tanh(gA);
            const float h = so * fast_tanh(c_state);
            h_lds[nxt + wave * 4 + jl] = (_Float16)h;   // next-iteration B source
            out[(size_t)b * HDIM + wave * 4 + jl] = h;  // hs[:, 11, :]
        }
        __syncthreads();                        // single barrier per step
    }
}

extern "C" void kernel_launch(void* const* d_in, const int* in_sizes, int n_in,
                              void* d_out, int out_size, void* d_ws, size_t ws_size,
                              hipStream_t stream) {
    const float* one_hot = (const float*)d_in[0];   // [B, N]
    const float* gmat    = (const float*)d_in[1];   // [N, N]
    const float* W_ih    = (const float*)d_in[2];   // [4H, N]
    const float* W_hh    = (const float*)d_in[3];   // [4H, H]
    float* out = (float*)d_out;                     // [B, H]
    unsigned char* seq8 = (unsigned char*)d_ws;     // [B] final symbols

    grammar_expand_kernel<<<BATCH / 256, 256, 0, stream>>>(one_hot, gmat, seq8);
    lstm_scan_kernel<<<1, THREADS, 0, stream>>>(seq8, W_ih, W_hh, out);
}
